// Seq2SeqLSTM_66254165508237
// MI455X (gfx1250) — matrix operand order repaired and checked by reading, not compiled, not run on hardware
//
#include <hip/hip_runtime.h>
#include <hip/hip_bf16.h>

typedef __attribute__((ext_vector_type(16))) _Float16 v16h;
typedef __attribute__((ext_vector_type(8)))  _Float16 v8h;
typedef __attribute__((ext_vector_type(8)))  float    v8f;

// Problem dims
#define BB 64
#define SS 64
#define TT 32
#define EE 256
#define HH 256
#define VV 32000
#define LL 2

__device__ __forceinline__ float sigm(float x) { return 1.0f / (1.0f + __expf(-x)); }

// A-operand fragment (16x32 f16, wave32 layout, ISA 7.12.2):
// element e: K = kb + 16*(e>>3) + 8*khalf + (e&7)  -> two contiguous 8-half runs
__device__ __forceinline__ v16h load_afrag_h(const _Float16* __restrict__ row, int kb, int half) {
    const v8h lo = *(const v8h*)(row + kb + 8 * half);
    const v8h hi = *(const v8h*)(row + kb + 16 + 8 * half);
    return __builtin_shufflevector(lo, hi, 0, 1, 2, 3, 4, 5, 6, 7, 8, 9, 10, 11, 12, 13, 14, 15);
}

// Same fragment but gathered from an f32 row (embedding) with on-the-fly convert.
__device__ __forceinline__ v16h load_afrag_f32(const float* __restrict__ row, int kb, int half) {
    v16h r;
    const float* p0 = row + kb + 8 * half;
    const float* p1 = row + kb + 16 + 8 * half;
#pragma unroll
    for (int i = 0; i < 8; ++i) {
        r[i]     = (_Float16)p0[i];
        r[8 + i] = (_Float16)p1[i];
    }
    return r;
}

// Agent-scope grid sync across the 16 blocks of one step kernel.
// cnt is monotonically increasing within one kernel_launch call (re-zeroed by
// init_kernel each call); target = 16 * arrival_epoch.
__device__ __forceinline__ void grid_sync(int* cnt, int target) {
    __threadfence();   // release prior global stores (h0) to agent scope
    __syncthreads();
    if (threadIdx.x == 0) {
        __hip_atomic_fetch_add(cnt, 1, __ATOMIC_RELEASE, __HIP_MEMORY_SCOPE_AGENT);
        while (__hip_atomic_load(cnt, __ATOMIC_ACQUIRE, __HIP_MEMORY_SCOPE_AGENT) < target)
            __builtin_amdgcn_s_sleep(1);
    }
    __syncthreads();
    __threadfence();   // acquire: see other blocks' h0 stores
}

// ---------------------------------------------------------------------------
// One LSTM layer, one (mtile, ntile) group per wave, all 4 gates in-register.
// A fragment is loaded once per k-step and feeds 4 WMMAs (B at 128KB strides).
// Gate nonlinearities + c/h update fused, no LDS.
// EMB: A rows gathered from f32 embedding table; else from f16 h matrix.
// ---------------------------------------------------------------------------
template <bool EMB>
__device__ __forceinline__ void cell_group(
    const float* __restrict__ erow,      // EMB: emb row for this lane's batch row
    const _Float16* __restrict__ xrow,   // !EMB: f16 input row for this lane
    const _Float16* __restrict__ hprevrow,
    const _Float16* __restrict__ Wih,    // [1024][256] layer slice
    const _Float16* __restrict__ Whh,    // [1024][256]
    const float* __restrict__ bias,      // [1024]
    float* __restrict__ cst,             // [64][256] layer slice
    _Float16* __restrict__ hout,         // [64][256]
    int mt, int ntile, int r, int half)
{
    const int ncol = ntile * 16 + r;          // column within H (B-tile column)
    v8f acc[4] = {{}, {}, {}, {}};            // i, f, g, o tiles

#pragma unroll
    for (int kb = 0; kb < EE; kb += 32) {
        v16h a = EMB ? load_afrag_f32(erow, kb, half) : load_afrag_h(xrow, kb, half);
#pragma unroll
        for (int g = 0; g < 4; ++g) {
            v16h b = load_afrag_h(Wih + ((size_t)(g * 256 + ncol)) * EE, kb, half);
            acc[g] = __builtin_amdgcn_wmma_f32_16x16x32_f16(false, a, false, b, (short)0,
                                                            acc[g], false, false);
        }
    }
#pragma unroll
    for (int kb = 0; kb < HH; kb += 32) {
        v16h a = load_afrag_h(hprevrow, kb, half);
#pragma unroll
        for (int g = 0; g < 4; ++g) {
            v16h b = load_afrag_h(Whh + ((size_t)(g * 256 + ncol)) * HH, kb, half);
            acc[g] = __builtin_amdgcn_wmma_f32_16x16x32_f16(false, a, false, b, (short)0,
                                                            acc[g], false, false);
        }
    }

    const float bi = bias[0 * 256 + ncol];
    const float bf = bias[1 * 256 + ncol];
    const float bg = bias[2 * 256 + ncol];
    const float bo = bias[3 * 256 + ncol];

    // C/D layout: element j -> batch row mt*16 + j + 8*half, col ncol.
#pragma unroll
    for (int j = 0; j < 8; ++j) {
        int brow = mt * 16 + j + 8 * half;
        int idx  = brow * HH + ncol;
        float gi = acc[0][j] + bi;
        float gf = acc[1][j] + bf;
        float gg = acc[2][j] + bg;
        float go = acc[3][j] + bo;
        float cn = sigm(gf) * cst[idx] + sigm(gi) * tanhf(gg);
        cst[idx]  = cn;
        hout[idx] = (_Float16)(sigm(go) * tanhf(cn));
    }
}

// ---------------------------------------------------------------------------
// One full stacked-LSTM timestep (layer0 with fused embedding gather, grid
// sync, layer1). grid = 16 blocks x 128 threads (4 waves); 64 waves = 64
// (mtile, ntile) groups per layer.
// ---------------------------------------------------------------------------
__global__ void __launch_bounds__(128)
lstm_step_kernel(const float* __restrict__ emb,  // [V][256]
                 const int* __restrict__ toks, int tstride, int toff,
                 const _Float16* __restrict__ h0prev, _Float16* __restrict__ h0out,
                 float* __restrict__ c0,
                 const _Float16* __restrict__ Wih0, const _Float16* __restrict__ Whh0,
                 const float* __restrict__ b0,
                 const _Float16* __restrict__ h1prev, _Float16* __restrict__ h1out,
                 float* __restrict__ c1,
                 const _Float16* __restrict__ Wih1, const _Float16* __restrict__ Whh1,
                 const float* __restrict__ b1,
                 int* __restrict__ sync_cnt, int sync_target)
{
    const int lane  = threadIdx.x & 31;
    const int wave  = threadIdx.x >> 5;
    const int r     = lane & 15;
    const int half  = lane >> 4;
    const int group = blockIdx.x * 4 + wave;  // 0..63
    const int ntile = group >> 2;             // 0..15
    const int mt    = group & 3;              // 0..3
    const int m     = mt * 16 + r;            // batch row for A fragment

    // Layer 0: input is always an embedding gather.
    const int tok = toks[m * tstride + toff];
    const float* erow = emb + (size_t)tok * EE;
    cell_group<true>(erow, nullptr, h0prev + (size_t)m * HH,
                     Wih0, Whh0, b0, c0, h0out, mt, ntile, r, half);

    grid_sync(sync_cnt, sync_target);

    // Layer 1: input is the freshly written layer-0 h.
    cell_group<false>(nullptr, h0out + (size_t)m * EE, h1prev + (size_t)m * HH,
                      Wih1, Whh1, b1, c1, h1out, mt, ntile, r, half);
}

// ---------------------------------------------------------------------------
// Logits: pred = h1 @ Wout^T + b_out -> out[b][t][v].
// One wave per 16-wide vocab tile; B loaded once per k-step, 4 mtile accs.
// grid = 250 blocks x 256 threads (8 waves).
// ---------------------------------------------------------------------------
__global__ void __launch_bounds__(256)
logits_kernel(const _Float16* __restrict__ h16,     // [64][256]
              const _Float16* __restrict__ Wout16,  // [32000][256]
              const float* __restrict__ b_out,      // [32000]
              float* __restrict__ out,              // [64][31][32000]
              int t)
{
    const int lane  = threadIdx.x & 31;
    const int wave  = threadIdx.x >> 5;  // 0..7
    const int r     = lane & 15;
    const int half  = lane >> 4;
    const int ntile = blockIdx.x * 8 + wave;  // 0..1999
    const int n     = ntile * 16 + r;         // vocab column

    v8f acc[4] = {{}, {}, {}, {}};
#pragma unroll
    for (int kb = 0; kb < HH; kb += 32) {
        v16h b = load_afrag_h(Wout16 + (size_t)n * HH, kb, half);
#pragma unroll
        for (int mt = 0; mt < 4; ++mt) {
            v16h a = load_afrag_h(h16 + (size_t)(mt * 16 + r) * HH, kb, half);
            acc[mt] = __builtin_amdgcn_wmma_f32_16x16x32_f16(false, a, false, b, (short)0,
                                                             acc[mt], false, false);
        }
    }
    const float bv = b_out[n];
#pragma unroll
    for (int mt = 0; mt < 4; ++mt)
#pragma unroll
        for (int j = 0; j < 8; ++j) {
            int brow = mt * 16 + j + 8 * half;
            out[((size_t)brow * (TT - 1) + t) * VV + n] = acc[mt][j] + bv;
        }
}

// ---------------------------------------------------------------------------
// Greedy argmax (first occurrence of max, like jnp.argmax) -> buf[:, t+1]
// ---------------------------------------------------------------------------
__global__ void __launch_bounds__(256)
argmax_kernel(const float* __restrict__ out, int t, int* __restrict__ buf)
{
    __shared__ float sv[256];
    __shared__ int   si[256];
    const int b   = blockIdx.x;
    const int tid = threadIdx.x;
    const float* row = out + ((size_t)b * (TT - 1) + t) * VV;
    float best = -3.402823466e38f;
    int   bi   = 0;
    for (int v = tid; v < VV; v += 256) {
        float x = row[v];
        if (x > best) { best = x; bi = v; }
    }
    sv[tid] = best; si[tid] = bi;
    __syncthreads();
    for (int s = 128; s > 0; s >>= 1) {
        if (tid < s) {
            if (sv[tid + s] > sv[tid] ||
                (sv[tid + s] == sv[tid] && si[tid + s] < si[tid])) {
                sv[tid] = sv[tid + s];
                si[tid] = si[tid + s];
            }
        }
        __syncthreads();
    }
    if (tid == 0) buf[b * TT + t + 1] = si[0];
}

__global__ void cvt_kernel(const float* __restrict__ src, _Float16* __restrict__ dst, int n)
{
    int i  = blockIdx.x * blockDim.x + threadIdx.x;
    int st = gridDim.x * blockDim.x;
    for (; i < n; i += st) dst[i] = (_Float16)src[i];
}

__global__ void init_kernel(float* __restrict__ cst, _Float16* __restrict__ h16,
                            const int* __restrict__ target, int* __restrict__ buf,
                            int* __restrict__ sync_cnt)
{
    int i = blockIdx.x * blockDim.x + threadIdx.x;
    if (i < LL * BB * HH) cst[i] = 0.0f;
    if (i < LL * 2 * BB * HH) h16[i] = (_Float16)0.0f;
    if (i < BB) buf[i * TT] = target[i * TT];  // BOS = target_seq[:,0]
    if (i == 0) *sync_cnt = 0;
}

extern "C" void kernel_launch(void* const* d_in, const int* in_sizes, int n_in,
                              void* d_out, int out_size, void* d_ws, size_t ws_size,
                              hipStream_t stream) {
    (void)in_sizes; (void)n_in; (void)out_size; (void)ws_size;

    const int*   input_seq  = (const int*)d_in[0];    // [64][64]
    const int*   target_seq = (const int*)d_in[1];    // [64][32]
    const float* enc_emb    = (const float*)d_in[2];  // [32000][256]
    const float* dec_emb    = (const float*)d_in[3];
    const float* enc_Wih    = (const float*)d_in[4];  // [2][1024][256]
    const float* enc_Whh    = (const float*)d_in[5];
    const float* enc_b      = (const float*)d_in[6];  // [2][1024]
    const float* dec_Wih    = (const float*)d_in[7];
    const float* dec_Whh    = (const float*)d_in[8];
    const float* dec_b      = (const float*)d_in[9];
    const float* W_out      = (const float*)d_in[10]; // [32000][256]
    const float* b_out      = (const float*)d_in[11]; // [32000]
    float*       out        = (float*)d_out;          // [64][31][32000]

    // Workspace carve (256B aligned regions), ~20.5 MB total.
    char* w = (char*)d_ws;
    auto take = [&](size_t bytes) {
        char* p = w;
        w += (bytes + 255) & ~(size_t)255;
        return p;
    };
    const size_t WSZ = (size_t)LL * 4 * HH * EE;  // 2*1024*256 elems per weight tensor
    _Float16* encWih16 = (_Float16*)take(WSZ * 2);
    _Float16* encWhh16 = (_Float16*)take(WSZ * 2);
    _Float16* decWih16 = (_Float16*)take(WSZ * 2);
    _Float16* decWhh16 = (_Float16*)take(WSZ * 2);
    _Float16* Wout16   = (_Float16*)take((size_t)VV * HH * 2);
    float*    cst      = (float*)take((size_t)LL * BB * HH * 4);
    _Float16* h16      = (_Float16*)take((size_t)LL * 2 * BB * HH * 2);  // [l][pp][64][256]
    int*      buf      = (int*)take((size_t)BB * TT * 4);
    int*      sync_cnt = (int*)take(256);

    // Per-call preprocessing (idempotent, graph-capture safe).
    cvt_kernel<<<512, 256, 0, stream>>>(enc_Wih, encWih16, (int)WSZ);
    cvt_kernel<<<512, 256, 0, stream>>>(enc_Whh, encWhh16, (int)WSZ);
    cvt_kernel<<<512, 256, 0, stream>>>(dec_Wih, decWih16, (int)WSZ);
    cvt_kernel<<<512, 256, 0, stream>>>(dec_Whh, decWhh16, (int)WSZ);
    cvt_kernel<<<2048, 256, 0, stream>>>(W_out, Wout16, VV * HH);
    init_kernel<<<256, 256, 0, stream>>>(cst, h16, target_seq, buf, sync_cnt);

    int pp = 0;        // ping-pong parity for both layers' h buffers
    int epoch = 0;     // grid-sync arrival epoch (16 blocks per step)
    auto hb = [&](int l, int p) { return h16 + ((size_t)l * 2 + p) * BB * HH; };

    auto step = [&](const float* emb, const int* toks, int tstride, int toff,
                    const _Float16* Wih, const _Float16* Whh, const float* bb) {
        ++epoch;
        lstm_step_kernel<<<16, 128, 0, stream>>>(
            emb, toks, tstride, toff,
            hb(0, pp), hb(0, pp ^ 1), cst,
            Wih, Whh, bb,
            hb(1, pp), hb(1, pp ^ 1), cst + (size_t)BB * HH,
            Wih + WSZ / 2, Whh + WSZ / 2, bb + 4 * HH,
            sync_cnt, 16 * epoch);
        pp ^= 1;
    };

    // ---- Encoder: 64 steps ----
    for (int t = 0; t < SS; ++t)
        step(enc_emb, input_seq, SS, t, encWih16, encWhh16, enc_b);

    // ---- Decoder: 31 outer steps; masked inner positions p>t are exact no-ops ----
    for (int t = 0; t < TT - 1; ++t) {
        for (int p = 0; p <= t; ++p)
            step(dec_emb, buf, TT, p, decWih16, decWhh16, dec_b);
        logits_kernel<<<VV / (16 * 8), 256, 0, stream>>>(hb(1, pp), Wout16, b_out, out, t);
        argmax_kernel<<<BB, 256, 0, stream>>>(out, t, buf);
    }
}
